// SwiGLUFFN_27238682591327
// MI455X (gfx1250) — 8-bit matrix operand order repaired and checked by reading, compiled, not run on hardware
//
#include <hip/hip_runtime.h>
#include <hip/hip_bf16.h>
#include <stdint.h>

typedef __attribute__((ext_vector_type(8))) int v8i;
typedef __attribute__((ext_vector_type(4))) int v4i;
typedef __attribute__((ext_vector_type(2))) int v2i;

#define EPSQ 1e-5f

template <typename T>
__global__ __launch_bounds__(256)
void rmsq_kernel(const T* __restrict__ x, int8_t* __restrict__ q,
                 float* __restrict__ sdeq, int D) {
  __shared__ float red[256];
  const size_t row = blockIdx.x;
  const T* xr = x + row * (size_t)D;
  float ss = 0.f, amax = 0.f;
  for (int i = threadIdx.x; i < D; i += 256) {
    float v = (float)xr[i];
    ss += v * v;
    amax = fmaxf(amax, fabsf(v));
  }
  red[threadIdx.x] = ss;
  __syncthreads();
  for (int s = 128; s > 0; s >>= 1) {
    if ((int)threadIdx.x < s) red[threadIdx.x] += red[threadIdx.x + s];
    __syncthreads();
  }
  const float rms = rsqrtf(red[0] / (float)D + 1e-6f);
  __syncthreads();
  red[threadIdx.x] = amax;
  __syncthreads();
  for (int s = 128; s > 0; s >>= 1) {
    if ((int)threadIdx.x < s) red[threadIdx.x] = fmaxf(red[threadIdx.x], red[threadIdx.x + s]);
    __syncthreads();
  }
  const float amax_n = red[0] * rms;
  const float clipped = fmaxf(amax_n, EPSQ);
  const float scale = 127.f / clipped;
  int8_t* qr = q + row * (size_t)D;
  for (int i = threadIdx.x; i < D; i += 256) {
    int qi = (int)rintf((float)xr[i] * rms * scale);
    qi = qi > 127 ? 127 : (qi < -128 ? -128 : qi);
    qr[i] = (int8_t)qi;
  }
  if (threadIdx.x == 0) sdeq[row] = clipped / 127.f;
}

__global__ __launch_bounds__(256)
void abssum1_kernel(const float* __restrict__ w, size_t n, float* __restrict__ part) {
  __shared__ float red[256];
  float s = 0.f;
  for (size_t i = (size_t)blockIdx.x * 256 + threadIdx.x; i < n; i += (size_t)gridDim.x * 256)
    s += fabsf(w[i]);
  red[threadIdx.x] = s;
  __syncthreads();
  for (int t = 128; t > 0; t >>= 1) {
    if ((int)threadIdx.x < t) red[threadIdx.x] += red[threadIdx.x + t];
    __syncthreads();
  }
  if (threadIdx.x == 0) part[blockIdx.x] = red[0];
}

__global__ __launch_bounds__(256)
void abssum2_kernel(const float* __restrict__ part, int n, float* __restrict__ out) {
  __shared__ float red[256];
  float s = 0.f;
  for (int i = threadIdx.x; i < n; i += 256) s += part[i];
  red[threadIdx.x] = s;
  __syncthreads();
  for (int t = 128; t > 0; t >>= 1) {
    if ((int)threadIdx.x < t) red[threadIdx.x] += red[threadIdx.x + t];
    __syncthreads();
  }
  if (threadIdx.x == 0) out[0] = red[0];
}

__global__ __launch_bounds__(256)
void wquant_kernel(const float* __restrict__ w, int8_t* __restrict__ q,
                   const float* __restrict__ sum, float invcnt, size_t n) {
  const float mean = sum[0] * invcnt;
  const float scale = 1.f / fmaxf(mean, EPSQ);
  for (size_t i = (size_t)blockIdx.x * blockDim.x + threadIdx.x; i < n;
       i += (size_t)gridDim.x * blockDim.x) {
    __builtin_prefetch(w + i + 4096, 0, 1);
    int qi = (int)rintf(w[i] * scale);
    qi = qi > 1 ? 1 : (qi < -1 ? -1 : qi);
    q[i] = (int8_t)qi;
  }
}

__device__ __forceinline__ v8i wmma_iu8(v8i a, v8i b, v8i c) {
  return __builtin_amdgcn_wmma_i32_16x16x64_iu8(true, a, true, b, c, false, false);
}

__device__ __forceinline__ void async_copy16(void* lds, const int8_t* gptr) {
  uint32_t ldsa = (uint32_t)(uintptr_t)lds;
  unsigned long long ga = (unsigned long long)(uintptr_t)gptr;
  asm volatile("global_load_async_to_lds_b128 %0, %1, off"
               :: "v"(ldsa), "v"(ga) : "memory");
}
__device__ __forceinline__ void wait_async_le(int n) {
  if (n == 0)      asm volatile("s_wait_asynccnt 0x0" ::: "memory");
  else if (n == 5) asm volatile("s_wait_asynccnt 0x5" ::: "memory");
  else             asm volatile("s_wait_asynccnt 0x6" ::: "memory");
}

__device__ __forceinline__ v8i load_a16x64(const int8_t* __restrict__ row, int k0, int half) {
  v8i a;
#pragma unroll
  for (int p = 0; p < 4; ++p) {
    v2i t = *(const v2i*)(row + k0 + p * 16 + half * 8);
    a[2 * p] = t[0];
    a[2 * p + 1] = t[1];
  }
  return a;
}

__device__ __forceinline__ v8i load_b64x16(const int8_t* __restrict__ row, int k0, int half) {
  v8i b;
#pragma unroll
  for (int p = 0; p < 4; ++p) {
    v2i t = *(const v2i*)(row + k0 + p * 16 + half * 8);
    b[2 * p] = t[0];
    b[2 * p + 1] = t[1];
  }
  return b;
}

#define KBLK   128
#define PITCH  144
#define A_BYTES (128 * PITCH)
#define B_BYTES (32 * PITCH)

__global__ __launch_bounds__(256)
void gemm1_kernel(const int8_t* __restrict__ xq, const float* __restrict__ sx,
                  const int8_t* __restrict__ w1q, const int8_t* __restrict__ w2q,
                  const float* __restrict__ sums, _Float16* __restrict__ h) {
  const int K = 2048, H = 5632;
  __shared__ int8_t smem[2][A_BYTES + 2 * B_BYTES];

  const int tid = threadIdx.x;
  const int lane = tid & 31, wv = tid >> 5;
  const int half = lane >> 4, l16 = lane & 15;
  const int wm = wv & 3, wn = wv >> 2;
  const int m_base = (int)blockIdx.y * 128;
  const int n_base = (int)blockIdx.x * 32;

  auto issue_stage = [&](int k0, int buf) {
    int8_t* base = smem[buf];
#pragma unroll
    for (int j = 0; j < 4; ++j) {
      int c = tid + j * 256;
      int row = c >> 3, col = (c & 7) * 16;
      async_copy16(base + row * PITCH + col,
                   xq + (size_t)(m_base + row) * K + k0 + col);
    }
    {
      int row = tid >> 3, col = (tid & 7) * 16;
      async_copy16(base + A_BYTES + row * PITCH + col,
                   w1q + (size_t)(n_base + row) * K + k0 + col);
      async_copy16(base + A_BYTES + B_BYTES + row * PITCH + col,
                   w2q + (size_t)(n_base + row) * K + k0 + col);
    }
  };

  const v8i z = {0, 0, 0, 0, 0, 0, 0, 0};
  v8i accg[2] = {z, z};
  v8i accu[2] = {z, z};

  issue_stage(0, 0);
  const int NSTAGE = K / KBLK;
  for (int s = 0; s < NSTAGE; ++s) {
    if (s + 1 < NSTAGE) {
      issue_stage((s + 1) * KBLK, (s + 1) & 1);
      wait_async_le(6);
    } else {
      wait_async_le(0);
    }
    __syncthreads();
    const int8_t* base = smem[s & 1];
    const int8_t* Arow0 = base + (wm * 32 + l16) * PITCH;
    const int8_t* Arow1 = base + (wm * 32 + 16 + l16) * PITCH;
    const int8_t* B1row = base + A_BYTES + (wn * 16 + l16) * PITCH;
    const int8_t* B2row = base + A_BYTES + B_BYTES + (wn * 16 + l16) * PITCH;
#pragma unroll
    for (int kk = 0; kk < KBLK; kk += 64) {
      v8i a0 = load_a16x64(Arow0, kk, half);
      v8i a1 = load_a16x64(Arow1, kk, half);
      v8i b1 = load_b64x16(B1row, kk, half);
      v8i b2 = load_b64x16(B2row, kk, half);
      accg[0] = wmma_iu8(a0, b1, accg[0]);
      accg[1] = wmma_iu8(a1, b1, accg[1]);
      accu[0] = wmma_iu8(a0, b2, accu[0]);
      accu[1] = wmma_iu8(a1, b2, accu[1]);
    }
    __syncthreads();
  }

  const float invcnt = 1.0f / (float)(5632 * 2048);
  const float sw1 = fmaxf(sums[0] * invcnt, EPSQ);
  const float sw2 = fmaxf(sums[1] * invcnt, EPSQ);
  const int col = n_base + wn * 16 + l16;
#pragma unroll
  for (int t = 0; t < 2; ++t) {
#pragma unroll
    for (int r = 0; r < 8; ++r) {
      const int m = m_base + wm * 32 + t * 16 + r + (half << 3);
      const float sA = sx[m];
      const float g = (float)accg[t][r] * sA * sw1;
      const float u = (float)accu[t][r] * sA * sw2;
      const float gate = g / (1.0f + __expf(-g));
      h[(size_t)m * H + col] = (_Float16)(gate * u);
    }
  }
}

__global__ __launch_bounds__(256)
void gemm2_kernel(const int8_t* __restrict__ hq, const float* __restrict__ sh,
                  const int8_t* __restrict__ w3q, const float* __restrict__ sums,
                  float* __restrict__ out) {
  const int K = 5632, NOUT = 2048;
  __shared__ int8_t smem[2][A_BYTES + B_BYTES];

  const int tid = threadIdx.x;
  const int lane = tid & 31, wv = tid >> 5;
  const int half = lane >> 4, l16 = lane & 15;
  const int wm = wv & 3, wn = wv >> 2;
  const int m_base = (int)blockIdx.y * 128;
  const int n_base = (int)blockIdx.x * 32;

  auto issue_stage = [&](int k0, int buf) {
    int8_t* base = smem[buf];
#pragma unroll
    for (int j = 0; j < 4; ++j) {
      int c = tid + j * 256;
      int row = c >> 3, col = (c & 7) * 16;
      async_copy16(base + row * PITCH + col,
                   hq + (size_t)(m_base + row) * K + k0 + col);
    }
    {
      int row = tid >> 3, col = (tid & 7) * 16;
      async_copy16(base + A_BYTES + row * PITCH + col,
                   w3q + (size_t)(n_base + row) * K + k0 + col);
    }
  };

  const v8i z = {0, 0, 0, 0, 0, 0, 0, 0};
  v8i acc[2] = {z, z};

  issue_stage(0, 0);
  const int NSTAGE = K / KBLK;
  for (int s = 0; s < NSTAGE; ++s) {
    if (s + 1 < NSTAGE) {
      issue_stage((s + 1) * KBLK, (s + 1) & 1);
      wait_async_le(5);
    } else {
      wait_async_le(0);
    }
    __syncthreads();
    const int8_t* base = smem[s & 1];
    const int8_t* Arow0 = base + (wm * 32 + l16) * PITCH;
    const int8_t* Arow1 = base + (wm * 32 + 16 + l16) * PITCH;
    const int8_t* Brow = base + A_BYTES + (wn * 16 + l16) * PITCH;
#pragma unroll
    for (int kk = 0; kk < KBLK; kk += 64) {
      v8i a0 = load_a16x64(Arow0, kk, half);
      v8i a1 = load_a16x64(Arow1, kk, half);
      v8i b = load_b64x16(Brow, kk, half);
      acc[0] = wmma_iu8(a0, b, acc[0]);
      acc[1] = wmma_iu8(a1, b, acc[1]);
    }
    __syncthreads();
  }

  const float invcnt = 1.0f / (float)(2048 * 5632);
  const float sw3 = fmaxf(sums[2] * invcnt, EPSQ);
  const int col = n_base + wn * 16 + l16;
#pragma unroll
  for (int t = 0; t < 2; ++t) {
#pragma unroll
    for (int r = 0; r < 8; ++r) {
      const int m = m_base + wm * 32 + t * 16 + r + (half << 3);
      out[(size_t)m * NOUT + col] = (float)acc[t][r] * sh[m] * sw3;
    }
  }
}

extern "C" void kernel_launch(void* const* d_in, const int* in_sizes, int n_in,
                              void* d_out, int out_size, void* d_ws, size_t ws_size,
                              hipStream_t stream) {
  (void)in_sizes; (void)n_in; (void)out_size; (void)ws_size;

  const int NT = 4 * 4096;
  const int D = 2048;
  const int H = 5632;
  const size_t HD = (size_t)H * D;

  const float* x  = (const float*)d_in[0];
  const float* w1 = (const float*)d_in[1];
  const float* w2 = (const float*)d_in[2];
  const float* w3 = (const float*)d_in[3];
  float* out = (float*)d_out;

  char* ws = (char*)d_ws;
  float* partials = (float*)ws;
  float* sums     = (float*)(ws + 4096);
  size_t off = 4352;
  int8_t*    xq  = (int8_t*)(ws + off);    off += (size_t)NT * D;
  float*     sx  = (float*)(ws + off);     off += (size_t)NT * 4;
  int8_t*    w1q = (int8_t*)(ws + off);    off += HD;
  int8_t*    w2q = (int8_t*)(ws + off);    off += HD;
  int8_t*    w3q = (int8_t*)(ws + off);    off += HD;
  _Float16*  h   = (_Float16*)(ws + off);  off += (size_t)NT * H * 2;
  int8_t*    hq  = (int8_t*)(ws + off);    off += (size_t)NT * H;
  float*     sh  = (float*)(ws + off);     off += (size_t)NT * 4;

  const float invHD = 1.0f / (float)HD;

  rmsq_kernel<float><<<NT, 256, 0, stream>>>(x, xq, sx, D);

  abssum1_kernel<<<1024, 256, 0, stream>>>(w1, HD, partials);
  abssum2_kernel<<<1, 256, 0, stream>>>(partials, 1024, sums + 0);
  abssum1_kernel<<<1024, 256, 0, stream>>>(w2, HD, partials);
  abssum2_kernel<<<1, 256, 0, stream>>>(partials, 1024, sums + 1);
  abssum1_kernel<<<1024, 256, 0, stream>>>(w3, HD, partials);
  abssum2_kernel<<<1, 256, 0, stream>>>(partials, 1024, sums + 2);
  wquant_kernel<<<2048, 256, 0, stream>>>(w1, w1q, sums + 0, invHD, HD);
  wquant_kernel<<<2048, 256, 0, stream>>>(w2, w2q, sums + 1, invHD, HD);
  wquant_kernel<<<2048, 256, 0, stream>>>(w3, w3q, sums + 2, invHD, HD);

  dim3 g1(H / 32, NT / 128);
  gemm1_kernel<<<g1, 256, 0, stream>>>(xq, sx, w1q, w2q, sums, h);

  rmsq_kernel<_Float16><<<NT, 256, 0, stream>>>(h, hq, sh, H);

  dim3 g2(D / 32, NT / 128);
  gemm2_kernel<<<g2, 256, 0, stream>>>(hq, sh, w3q, sums, out);
}
